// Model_28948079575344
// MI455X (gfx1250) — compile-verified, matrix operand order checked by reading, not run on hardware
//
#include <hip/hip_runtime.h>
#include <hip/hip_bf16.h>

typedef __attribute__((ext_vector_type(16))) _Float16 v16h;
typedef __attribute__((ext_vector_type(8)))  float    v8f;

#define GAT_N 20000
#define GAT_E 320000
#define GAT_FIN 128
#define GAT_D 256
#define GAT_HEADS 4
#define GAT_HID 64
#define NEG_SLOPE 0.2f
#define NGRAPH 16
#define KP_CAT 288           // 257 padded up to multiple of 32
#define ENC_NEG_INF 0x007FFFFFu   // enc(-inf)

static __device__ __forceinline__ unsigned enc_f(float f) {
    unsigned u = __float_as_uint(f);
    return (u & 0x80000000u) ? ~u : (u | 0x80000000u);
}
static __device__ __forceinline__ float dec_f(unsigned u) {
    unsigned b = (u & 0x80000000u) ? (u & 0x7FFFFFFFu) : ~u;
    return __uint_as_float(b);
}

// ---------------------------------------------------------------------------
// WMMA GEMM with register-level A reuse:
//   C[M x ncols] = A16[M x lda] * Bt16[ncols x lda]  (+bias, optional relu)
// Bt is B transposed (row = output column); both f16 row-major; K multiple of
// 32. Each wave computes a 16(M) x 64(N) strip: one A fragment feeds four
// WMMAs against four B fragments (4x less A traffic than tile-per-wave).
// 128 threads = 4 waves -> block covers 16 x 256.
// grid.x = M/16, grid.y = ceil(ncols/256).
// ---------------------------------------------------------------------------
__global__ void wmma_gemm(const _Float16* __restrict__ A,
                          const _Float16* __restrict__ Bt,
                          float* __restrict__ C,
                          const float* __restrict__ bias,
                          int lda, int K, int ncols, int ldc, int ncols_real,
                          int do_relu)
{
    const int wave = threadIdx.x >> 5;
    const int lane = threadIdx.x & 31;
    const int hl   = lane >> 4;      // half-wave select
    const int l16  = lane & 15;
    const int m0 = blockIdx.x * 16;
    const int n0 = (blockIdx.y * 4 + wave) * 64;
    if (n0 >= ncols) return;

    const _Float16* arow = A + (size_t)(m0 + l16) * lda;
    const _Float16* brow[4];
#pragma unroll
    for (int t = 0; t < 4; ++t) {
        int r = n0 + t * 16 + l16;
        if (r > ncols - 1) r = ncols - 1;  // clamp: keeps loads in-bounds,
        brow[t] = Bt + (size_t)r * lda;    // results discarded by store guard
    }

    v8f acc[4] = {};
    for (int k0 = 0; k0 < K; k0 += 32) {
        // A 16x32 f16 fragment: lanes 0-15 cover K=[0..7],[16..23];
        // lanes 16-31 cover K=[8..15],[24..31] (two 16B loads per lane).
        v16h a;
        ((float4*)&a)[0] = *(const float4*)(arow + k0 +      hl * 8);
        ((float4*)&a)[1] = *(const float4*)(arow + k0 + 16 + hl * 8);
        __builtin_prefetch(arow + k0 + 128, 0, 1);
#pragma unroll
        for (int t = 0; t < 4; ++t) {
            v16h b;
            ((float4*)&b)[0] = *(const float4*)(brow[t] + k0 +      hl * 8);
            ((float4*)&b)[1] = *(const float4*)(brow[t] + k0 + 16 + hl * 8);
            acc[t] = __builtin_amdgcn_wmma_f32_16x16x32_f16(
                false, a, false, b, (short)0, acc[t], false, false);
        }
    }

#pragma unroll
    for (int t = 0; t < 4; ++t) {
        const int n = n0 + t * 16 + l16;
        if (n < ncols_real) {
            const float bv = bias ? bias[n] : 0.0f;
#pragma unroll
            for (int r = 0; r < 8; ++r) {
                float v = acc[t][r] + bv;
                if (do_relu) v = fmaxf(v, 0.0f);
                C[(size_t)(m0 + r + hl * 8) * ldc + n] = v;
            }
        }
    }
}

// ---------------------------------------------------------------------------
// Conversions / fills
// ---------------------------------------------------------------------------
__global__ void fill_u32(unsigned* p, unsigned v, int total) {
    int i = blockIdx.x * blockDim.x + threadIdx.x;
    if (i < total) p[i] = v;
}

// f32 [rows x incols] -> f16 [rows x outld], zero-padded columns
__global__ void cvt_a16(const float* __restrict__ in, _Float16* __restrict__ out,
                        int incols, int outld, int total) {
    int i = blockIdx.x * blockDim.x + threadIdx.x;
    if (i >= total) return;
    int r = i / outld, c = i - r * outld;
    out[i] = (_Float16)(c < incols ? in[(size_t)r * incols + c] : 0.0f);
}

// W f32 [K x ncols] -> Wt f16 [npad x kp] transposed + zero padded
__global__ void cvt_w_t(const float* __restrict__ W, _Float16* __restrict__ Wt,
                        int K, int ncols, int kp, int total) {
    int i = blockIdx.x * blockDim.x + threadIdx.x;
    if (i >= total) return;
    int ncol = i / kp, k = i - ncol * kp;
    float v = (k < K && ncol < ncols) ? W[(size_t)k * ncols + ncol] : 0.0f;
    Wt[i] = (_Float16)v;
}

// [jk | gate | 0-pad] -> f16 [N x 288]
__global__ void concat_a16(const float* __restrict__ jk,
                           const float* __restrict__ gate,
                           _Float16* __restrict__ out, int total) {
    int i = blockIdx.x * blockDim.x + threadIdx.x;
    if (i >= total) return;
    int r = i / KP_CAT, c = i - r * KP_CAT;
    float v = (c < GAT_D) ? jk[(size_t)r * GAT_D + c] : (c == GAT_D ? gate[r] : 0.0f);
    out[i] = (_Float16)v;
}

// ---------------------------------------------------------------------------
// Attention pieces
// ---------------------------------------------------------------------------
// per (node, head): asn = a_s . h ; adn = a_d . h
__global__ void alpha_dots(const float* __restrict__ h,
                           const float* __restrict__ a_s,
                           const float* __restrict__ a_d,
                           float* __restrict__ asn, float* __restrict__ adn,
                           int total) {
    int i = blockIdx.x * blockDim.x + threadIdx.x;
    if (i >= total) return;
    int n = i >> 2, hd = i & 3;
    const float* hp  = h   + (size_t)n * GAT_D + hd * GAT_HID;
    const float* asp = a_s + hd * GAT_HID;
    const float* adp = a_d + hd * GAT_HID;
    float s1 = 0.f, s2 = 0.f;
#pragma unroll 8
    for (int d = 0; d < GAT_HID; ++d) { s1 += hp[d] * asp[d]; s2 += hp[d] * adp[d]; }
    asn[i] = s1; adn[i] = s2;
}

__device__ __forceinline__ float edge_logit(const float* asn, const float* adn,
                                            int s, int d, int hd) {
    float l = asn[s * GAT_HEADS + hd] + adn[d * GAT_HEADS + hd];
    return l > 0.f ? l : NEG_SLOPE * l;
}

// pass 1: segment max of leaky logits over dst (order-preserving uint encode)
__global__ void edge_max(const int* __restrict__ src, const int* __restrict__ dst,
                         const float* __restrict__ asn, const float* __restrict__ adn,
                         unsigned* __restrict__ menc, int total) {
    int i = blockIdx.x * blockDim.x + threadIdx.x;
    if (i >= total) return;
    int e = i >> 2, hd = i & 3;
    float l = edge_logit(asn, adn, src[e], dst[e], hd);
    atomicMax(&menc[dst[e] * GAT_HEADS + hd], enc_f(l));
}

// pass 2: segment sum of exp(l - m)
__global__ void edge_expsum(const int* __restrict__ src, const int* __restrict__ dst,
                            const float* __restrict__ asn, const float* __restrict__ adn,
                            const unsigned* __restrict__ menc, float* __restrict__ ssum,
                            int total) {
    int i = blockIdx.x * blockDim.x + threadIdx.x;
    if (i >= total) return;
    int e = i >> 2, hd = i & 3;
    int d = dst[e];
    float l = edge_logit(asn, adn, src[e], d, hd);
    float m = dec_f(menc[d * GAT_HEADS + hd]);
    if (m < -3.0e38f) m = 0.0f;  // empty-segment guard
    atomicAdd(&ssum[d * GAT_HEADS + hd], __expf(l - m));
}

// pass 3: agg[dst] += h[src] * alpha ; thread per (edge, 4-float chunk)
__global__ void edge_aggregate(const int* __restrict__ src, const int* __restrict__ dst,
                               const float* __restrict__ h,
                               const float* __restrict__ asn, const float* __restrict__ adn,
                               const unsigned* __restrict__ menc,
                               const float* __restrict__ ssum,
                               float* __restrict__ agg, int total) {
    int i = blockIdx.x * blockDim.x + threadIdx.x;
    if (i >= total) return;
    int e = i >> 6;
    int j = i & 63;          // 64 chunks of 4 floats over D=256
    int d0 = j * 4;
    int hd = d0 >> 6;        // head owning this chunk
    int s = src[e], dn = dst[e];
    float l = edge_logit(asn, adn, s, dn, hd);
    float m = dec_f(menc[dn * GAT_HEADS + hd]);
    if (m < -3.0e38f) m = 0.0f;
    float alpha = __expf(l - m) / (ssum[dn * GAT_HEADS + hd] + 1e-16f);
    const float* hp = h + (size_t)s * GAT_D + d0;
    float* op = agg + (size_t)dn * GAT_D + d0;
    atomicAdd(op + 0, hp[0] * alpha);
    atomicAdd(op + 1, hp[1] * alpha);
    atomicAdd(op + 2, hp[2] * alpha);
    atomicAdd(op + 3, hp[3] * alpha);
}

// bias + (optional) relu + running JK max; act also becomes next-layer input
__global__ void post_layer(const float* __restrict__ agg, const float* __restrict__ bias,
                           float* __restrict__ xnext, float* __restrict__ jk,
                           int do_relu, int jk_init, int total) {
    int i = blockIdx.x * blockDim.x + threadIdx.x;
    if (i >= total) return;
    float v = agg[i] + bias[i & (GAT_D - 1)];
    float act = do_relu ? fmaxf(v, 0.0f) : v;
    xnext[i] = act;
    jk[i] = jk_init ? act : fmaxf(jk[i], act);
}

// ---------------------------------------------------------------------------
// Gate head
// ---------------------------------------------------------------------------
__global__ void gate_logit(const float* __restrict__ g1, const float* __restrict__ Wg2,
                           const float* __restrict__ bg2, float* __restrict__ gl, int n) {
    int i = blockIdx.x * blockDim.x + threadIdx.x;
    if (i >= n) return;
    const float* gp = g1 + (size_t)i * GAT_HID;
    float s = bg2[0];
#pragma unroll 8
    for (int d = 0; d < GAT_HID; ++d) s += gp[d] * Wg2[d];
    gl[i] = s;
}

__global__ void batch_max(const float* __restrict__ gl, const int* __restrict__ batch,
                          unsigned* __restrict__ mg, int n) {
    int i = blockIdx.x * blockDim.x + threadIdx.x;
    if (i >= n) return;
    atomicMax(&mg[batch[i]], enc_f(gl[i]));
}
__global__ void batch_sum(const float* __restrict__ gl, const int* __restrict__ batch,
                          const unsigned* __restrict__ mg, float* __restrict__ sg, int n) {
    int i = blockIdx.x * blockDim.x + threadIdx.x;
    if (i >= n) return;
    int b = batch[i];
    float m = dec_f(mg[b]); if (m < -3.0e38f) m = 0.0f;
    atomicAdd(&sg[b], __expf(gl[i] - m));
}
__global__ void batch_div(const float* __restrict__ gl, const int* __restrict__ batch,
                          const unsigned* __restrict__ mg, const float* __restrict__ sg,
                          float* __restrict__ gate, int n) {
    int i = blockIdx.x * blockDim.x + threadIdx.x;
    if (i >= n) return;
    int b = batch[i];
    float m = dec_f(mg[b]); if (m < -3.0e38f) m = 0.0f;
    gate[i] = __expf(gl[i] - m) / (sg[b] + 1e-16f);
}

// ---------------------------------------------------------------------------
// Host-side orchestration
// ---------------------------------------------------------------------------
#define CDIV(a, b) (((a) + (b) - 1) / (b))

extern "C" void kernel_launch(void* const* d_in, const int* in_sizes, int n_in,
                              void* d_out, int out_size, void* d_ws, size_t ws_size,
                              hipStream_t stream) {
    const float* x      = (const float*)d_in[0];
    const int*   ei     = (const int*)d_in[1];
    const int*   batch  = (const int*)d_in[2];
    const int*   src    = ei;
    const int*   dst    = ei + GAT_E;
    const float* W[4]   = {(const float*)d_in[3],  (const float*)d_in[7],
                           (const float*)d_in[11], (const float*)d_in[15]};
    const float* AS[4]  = {(const float*)d_in[4],  (const float*)d_in[8],
                           (const float*)d_in[12], (const float*)d_in[16]};
    const float* AD[4]  = {(const float*)d_in[5],  (const float*)d_in[9],
                           (const float*)d_in[13], (const float*)d_in[17]};
    const float* B[4]   = {(const float*)d_in[6],  (const float*)d_in[10],
                           (const float*)d_in[14], (const float*)d_in[18]};
    const float* Wg1 = (const float*)d_in[19]; const float* bg1 = (const float*)d_in[20];
    const float* Wg2 = (const float*)d_in[21]; const float* bg2 = (const float*)d_in[22];
    const float* Wm1 = (const float*)d_in[23]; const float* bm1 = (const float*)d_in[24];
    const float* Wm2 = (const float*)d_in[25]; const float* bm2 = (const float*)d_in[26];
    const float* Wm3 = (const float*)d_in[27]; const float* bm3 = (const float*)d_in[28];
    float* out = (float*)d_out;

    // ---- carve workspace ----
    char* ws = (char*)d_ws;
    size_t off = 0;
    auto carve = [&](size_t bytes) -> void* {
        void* p = ws + off;
        off = (off + bytes + 255) & ~(size_t)255;
        return p;
    };
    const size_t ND = (size_t)GAT_N * GAT_D;
    float*     f_xcur = (float*)carve(ND * 4);            // current activations
    float*     f_h    = (float*)carve(ND * 4);            // xW (per-head features)
    float*     f_agg  = (float*)carve(ND * 4);            // attention aggregation
    float*     f_jk   = (float*)carve(ND * 4);            // running JK max
    _Float16*  h_act  = (_Float16*)carve((size_t)GAT_N * KP_CAT * 2); // f16 A operand
    _Float16*  h_w    = (_Float16*)carve((size_t)65536 * 2);          // f16 Bt operand
    float*     asn    = (float*)carve((size_t)GAT_N * GAT_HEADS * 4);
    float*     adn    = (float*)carve((size_t)GAT_N * GAT_HEADS * 4);
    unsigned*  menc   = (unsigned*)carve((size_t)GAT_N * GAT_HEADS * 4);
    float*     ssum   = (float*)carve((size_t)GAT_N * GAT_HEADS * 4);
    unsigned*  mg     = (unsigned*)carve(NGRAPH * 4);
    float*     sg     = (float*)carve(NGRAPH * 4);

    const int BT = 256;
    const int mtiles = GAT_N / 16;      // 1250, exact

    // ---- 4 GAT layers ----
    for (int l = 0; l < 4; ++l) {
        const int kin = (l == 0) ? GAT_FIN : GAT_D;
        // weight -> transposed f16 [256 x kin]
        {
            int tot = GAT_D * kin;
            cvt_w_t<<<CDIV(tot, BT), BT, 0, stream>>>(W[l], h_w, kin, GAT_D, kin, tot);
        }
        // activations -> f16 [N x kin]
        {
            int tot = GAT_N * kin;
            const float* ain = (l == 0) ? x : f_xcur;
            cvt_a16<<<CDIV(tot, BT), BT, 0, stream>>>(ain, h_act, kin, kin, tot);
        }
        // h = act @ W   (WMMA, fp32 accumulate)
        {
            dim3 g(mtiles, CDIV(GAT_D, 256));
            wmma_gemm<<<g, 128, 0, stream>>>(h_act, h_w, f_h, nullptr,
                                             kin, kin, GAT_D, GAT_D, GAT_D, 0);
        }
        // attention coefficients
        {
            int tot = GAT_N * GAT_HEADS;
            alpha_dots<<<CDIV(tot, BT), BT, 0, stream>>>(f_h, AS[l], AD[l], asn, adn, tot);
            fill_u32<<<CDIV(tot, BT), BT, 0, stream>>>(menc, ENC_NEG_INF, tot);
            fill_u32<<<CDIV(tot, BT), BT, 0, stream>>>((unsigned*)ssum, 0u, tot);
            fill_u32<<<CDIV((int)ND, BT), BT, 0, stream>>>((unsigned*)f_agg, 0u, (int)ND);
            int etot = GAT_E * GAT_HEADS;
            edge_max<<<CDIV(etot, BT), BT, 0, stream>>>(src, dst, asn, adn, menc, etot);
            edge_expsum<<<CDIV(etot, BT), BT, 0, stream>>>(src, dst, asn, adn, menc, ssum, etot);
            int atot = GAT_E * 64;
            edge_aggregate<<<CDIV(atot, BT), BT, 0, stream>>>(src, dst, f_h, asn, adn,
                                                              menc, ssum, f_agg, atot);
        }
        // bias + relu (layers 0-2) + JK running max; next-layer input in f_xcur
        post_layer<<<CDIV((int)ND, BT), BT, 0, stream>>>(f_agg, B[l], f_xcur, f_jk,
                                                         (l < 3) ? 1 : 0,
                                                         (l == 0) ? 1 : 0, (int)ND);
    }

    // ---- gate: relu(jk@Wg1+bg1)@Wg2+bg2, per-graph softmax ----
    {
        int tot = GAT_N * GAT_D;
        cvt_a16<<<CDIV(tot, BT), BT, 0, stream>>>(f_jk, h_act, GAT_D, GAT_D, tot);
        int wt = GAT_HID * GAT_D;
        cvt_w_t<<<CDIV(wt, BT), BT, 0, stream>>>(Wg1, h_w, GAT_D, GAT_HID, GAT_D, wt);
        dim3 g(mtiles, 1);
        wmma_gemm<<<g, 128, 0, stream>>>(h_act, h_w, f_h, bg1,
                                         GAT_D, GAT_D, GAT_HID, GAT_HID, GAT_HID, 1);
        float* gl   = asn;  // reuse
        float* gate = adn;  // reuse
        gate_logit<<<CDIV(GAT_N, BT), BT, 0, stream>>>(f_h, Wg2, bg2, gl, GAT_N);
        fill_u32<<<1, NGRAPH, 0, stream>>>(mg, ENC_NEG_INF, NGRAPH);
        fill_u32<<<1, NGRAPH, 0, stream>>>((unsigned*)sg, 0u, NGRAPH);
        batch_max<<<CDIV(GAT_N, BT), BT, 0, stream>>>(gl, batch, mg, GAT_N);
        batch_sum<<<CDIV(GAT_N, BT), BT, 0, stream>>>(gl, batch, mg, sg, GAT_N);
        batch_div<<<CDIV(GAT_N, BT), BT, 0, stream>>>(gl, batch, mg, sg, gate, GAT_N);

        // ---- MLP head ----
        int ct = GAT_N * KP_CAT;
        concat_a16<<<CDIV(ct, BT), BT, 0, stream>>>(f_jk, gate, h_act, ct);
        int w1 = 128 * KP_CAT;
        cvt_w_t<<<CDIV(w1, BT), BT, 0, stream>>>(Wm1, h_w, 257, 128, KP_CAT, w1);
        dim3 g1(mtiles, 1);
        wmma_gemm<<<g1, 128, 0, stream>>>(h_act, h_w, f_h, bm1,
                                          KP_CAT, KP_CAT, 128, 128, 128, 1);
        int t1 = GAT_N * 128;
        cvt_a16<<<CDIV(t1, BT), BT, 0, stream>>>(f_h, h_act, 128, 128, t1);
        int w2 = 64 * 128;
        cvt_w_t<<<CDIV(w2, BT), BT, 0, stream>>>(Wm2, h_w, 128, 64, 128, w2);
        dim3 g2(mtiles, 1);
        wmma_gemm<<<g2, 128, 0, stream>>>(h_act, h_w, f_agg, bm2,
                                          128, 128, 64, 64, 64, 1);
        int t2 = GAT_N * 64;
        cvt_a16<<<CDIV(t2, BT), BT, 0, stream>>>(f_agg, h_act, 64, 64, t2);
        int w3 = 16 * 64;   // pad 10 output cols -> 16
        cvt_w_t<<<CDIV(w3, BT), BT, 0, stream>>>(Wm3, h_w, 64, 10, 64, w3);
        dim3 g3(mtiles, 1);
        wmma_gemm<<<g3, 128, 0, stream>>>(h_act, h_w, out, bm3,
                                          64, 64, 16, 10, 10, 0);
    }
    (void)in_sizes; (void)n_in; (void)out_size; (void)ws_size;
}
